// TemporalSkip_63848983822722
// MI455X (gfx1250) — matrix operand order repaired and checked by reading, not compiled, not run on hardware
//
#include <hip/hip_runtime.h>
#include <hip/hip_bf16.h>
#include <math.h>

#define NNODES 100000
#define NEDGES 6400000
#define NTILES (NNODES / 16)   // 6250, exact

typedef __attribute__((ext_vector_type(16))) __bf16 v16bf;
typedef __attribute__((ext_vector_type(8)))  __bf16 v8bf;
typedef __attribute__((ext_vector_type(8)))  float  v8f;

// ---------------- WMMA helpers (CDNA5 16x16x32 bf16, wave32) ----------------

__device__ __forceinline__ v8f wmma_bf16(v16bf a, v16bf b, v8f c) {
  // 8 args: (neg_a, A, neg_b, B, c_mod, C, reuse_a, reuse_b)
  return __builtin_amdgcn_wmma_f32_16x16x32_bf16(false, a, false, b, (short)0, c,
                                                 false, false);
}

__device__ __forceinline__ v8f zero_v8f() {
  v8f z = {0.f, 0.f, 0.f, 0.f, 0.f, 0.f, 0.f, 0.f};
  return z;
}

__device__ __forceinline__ v16bf cvt16(float4 a, float4 b, float4 c, float4 d) {
  v16bf r;
  r[0]  = (__bf16)a.x; r[1]  = (__bf16)a.y; r[2]  = (__bf16)a.z; r[3]  = (__bf16)a.w;
  r[4]  = (__bf16)b.x; r[5]  = (__bf16)b.y; r[6]  = (__bf16)b.z; r[7]  = (__bf16)b.w;
  r[8]  = (__bf16)c.x; r[9]  = (__bf16)c.y; r[10] = (__bf16)c.z; r[11] = (__bf16)c.w;
  r[12] = (__bf16)d.x; r[13] = (__bf16)d.y; r[14] = (__bf16)d.z; r[15] = (__bf16)d.w;
  return r;
}

// A fragment: 16x32 tile of a row-major f32 matrix, converted to bf16.
// ISA layout: lanes 0-15 -> M=lane, K={0..7,16..23}; lanes 16-31 -> M=lane-16,
// K={8..15,24..31}.  Wide b128 loads (16B-aligned: k offsets are multiples of 8).
__device__ __forceinline__ v16bf load_a_f32(const float* __restrict__ base, int ld,
                                            int row0, int k0) {
  const int lane = threadIdx.x & 31;
  const int m = lane & 15;
  const int kb = (lane & 16) ? 8 : 0;
  const float* p = base + (size_t)(row0 + m) * ld + (k0 + kb);
  float4 a = *(const float4*)(p);
  float4 b = *(const float4*)(p + 4);
  float4 c = *(const float4*)(p + 16);
  float4 d = *(const float4*)(p + 20);
  return cvt16(a, b, c, d);
}

// Same A layout from a bf16 tile in LDS (row-major, stride ld): 2x ds_load_b128.
__device__ __forceinline__ v16bf load_a_lds(const __bf16* tile, int ld, int k0) {
  const int lane = threadIdx.x & 31;
  const int m = lane & 15;
  const int kb = (lane & 16) ? 8 : 0;
  const __bf16* p = tile + m * ld + k0 + kb;
  v8bf lo = *(const v8bf*)(p);
  v8bf hi = *(const v8bf*)(p + 16);
  return __builtin_shufflevector(lo, hi, 0, 1, 2, 3, 4, 5, 6, 7,
                                 8, 9, 10, 11, 12, 13, 14, 15);
}

// B fragment from pre-packed fragment-order bf16: one contiguous 32B per lane.
__device__ __forceinline__ v16bf load_b_packed(const __bf16* __restrict__ packed,
                                               int frag) {
  const int lane = threadIdx.x & 31;
  return *(const v16bf*)(packed + ((size_t)frag * 32 + lane) * 16);
}

// ---------------- weight pre-pack into B-fragment order ----------------
// For a KxNcols row-major f32 weight, fragment (kt,ct) covers K rows
// kt*32..kt*32+31 and columns ct*16..ct*16+15.  Lane n=lane&15 selects the
// column; lanes 0-15 hold K=kt*32+{0..7,16..23}, lanes 16-31 hold K=kt*32+{8..15,24..31}.
__global__ void __launch_bounds__(256) pack_b_kernel(const float* __restrict__ w,
                                                     int K, int Ncols,
                                                     __bf16* __restrict__ out) {
  int idx = blockIdx.x * blockDim.x + threadIdx.x;
  int ctn = Ncols >> 4;
  int nfrag = (K >> 5) * ctn;
  if (idx >= nfrag * 32) return;
  int frag = idx >> 5;
  int lane = idx & 31;
  int kt = frag / ctn;
  int ct = frag % ctn;
  int n = lane & 15;
  int kb = (lane & 16) ? 8 : 0;
  __bf16* o = out + (size_t)idx * 16;
#pragma unroll
  for (int j = 0; j < 16; ++j)
    o[j] = (__bf16)w[(size_t)(kt * 32 + kb + (j & 7) + (j >> 3) * 16) * Ncols + ct * 16 + n];
}

// ---------------- degree / normalization ----------------

__global__ void __launch_bounds__(256) deg_init_kernel(float* __restrict__ deg) {
  int i = blockIdx.x * blockDim.x + threadIdx.x;
  if (i < NNODES) deg[i] = 1.0f;  // self loop
}

__global__ void __launch_bounds__(256) deg_accum_kernel(const int* __restrict__ dst,
                                                        float* __restrict__ deg) {
  int stride = gridDim.x * blockDim.x;
  for (int e = blockIdx.x * blockDim.x + threadIdx.x; e < NEDGES; e += stride)
    atomicAdd(&deg[dst[e]], 1.0f);
}

__global__ void __launch_bounds__(256) dinv_kernel(const float* __restrict__ deg,
                                                   float* __restrict__ dinv) {
  int i = blockIdx.x * blockDim.x + threadIdx.x;
  if (i < NNODES) {
    float d = deg[i];
    dinv[i] = d > 0.f ? rsqrtf(d) : 0.f;
  }
}

// ---------------- encoder: h = tanh(tanh(x@W1+b1)@W2+b2) ----------------

__global__ void __launch_bounds__(256) encoder_kernel(
    const float* __restrict__ x,
    const __bf16* __restrict__ pw1, const float* __restrict__ b1,
    const __bf16* __restrict__ pw2, const float* __restrict__ b2,
    float* __restrict__ h) {
  __shared__ __bf16 lds[8][16 * 64];  // per-wave 16x64 intermediate tile
  const int wave = threadIdx.x >> 5;
  const int lane = threadIdx.x & 31;
  const int tile = blockIdx.x * 8 + wave;
  if (tile >= NTILES) return;  // wave-uniform exit; active waves keep full EXEC
  const int row0 = tile * 16;
  const int n = lane & 15;
  const int mbase = (lane & 16) ? 8 : 0;

  // GEMM1: [16x256] @ [256x64]
  v8f acc[4];
#pragma unroll
  for (int t = 0; t < 4; ++t) acc[t] = zero_v8f();
#pragma unroll
  for (int kt = 0; kt < 8; ++kt) {
    v16bf a = load_a_f32(x, 256, row0, kt * 32);
#pragma unroll
    for (int t = 0; t < 4; ++t) {
      v16bf b = load_b_packed(pw1, kt * 4 + t);
      acc[t] = wmma_bf16(a, b, acc[t]);
    }
  }
  // bias + tanh -> LDS (bf16, row-major 16x64), per-wave region (no barrier)
#pragma unroll
  for (int t = 0; t < 4; ++t) {
    const int col = t * 16 + n;
    const float bias = b1[col];
#pragma unroll
    for (int g = 0; g < 8; ++g)
      lds[wave][(mbase + g) * 64 + col] = (__bf16)tanhf(acc[t][g] + bias);
  }

  // GEMM2: [16x64] @ [64x32]
  v8f acc2[2];
#pragma unroll
  for (int t = 0; t < 2; ++t) acc2[t] = zero_v8f();
#pragma unroll
  for (int kt = 0; kt < 2; ++kt) {
    v16bf a = load_a_lds(&lds[wave][0], 64, kt * 32);
#pragma unroll
    for (int t = 0; t < 2; ++t) {
      v16bf b = load_b_packed(pw2, kt * 2 + t);
      acc2[t] = wmma_bf16(a, b, acc2[t]);
    }
  }
#pragma unroll
  for (int t = 0; t < 2; ++t) {
    const int col = t * 16 + n;
    const float bias = b2[col];
#pragma unroll
    for (int g = 0; g < 8; ++g)
      h[(size_t)(row0 + mbase + g) * 32 + col] = tanhf(acc2[t][g] + bias);
  }
}

// ---------------- GCN dense part (fused self-loop init) ----------------
// t = h @ W ;  agg = t * dinv^2

__global__ void __launch_bounds__(256) gcn_mm_kernel(const float* __restrict__ h,
                                                     const __bf16* __restrict__ pw,
                                                     const float* __restrict__ dinv,
                                                     float* __restrict__ t,
                                                     float* __restrict__ agg) {
  const int wave = threadIdx.x >> 5;
  const int lane = threadIdx.x & 31;
  const int tile = blockIdx.x * 8 + wave;
  if (tile >= NTILES) return;
  const int row0 = tile * 16;
  const int n = lane & 15;
  const int mbase = (lane & 16) ? 8 : 0;

  v16bf a = load_a_f32(h, 32, row0, 0);

  float di2[8];
#pragma unroll
  for (int g = 0; g < 8; ++g) {
    float di = dinv[row0 + mbase + g];
    di2[g] = di * di;
  }

#pragma unroll
  for (int ct = 0; ct < 2; ++ct) {
    v16bf b = load_b_packed(pw, ct);
    v8f acc = zero_v8f();
    acc = wmma_bf16(a, b, acc);
#pragma unroll
    for (int g = 0; g < 8; ++g) {
      const size_t o = (size_t)(row0 + mbase + g) * 32 + ct * 16 + n;
      const float v = acc[g];
      t[o] = v;
      agg[o] = v * di2[g];
    }
  }
}

// ---------------- GCN aggregation ----------------

// warp-per-edge, lane = feature: agg[dst] += t[src] * dinv[src]*dinv[dst]
__global__ void __launch_bounds__(256) edge_scatter_kernel(
    const int* __restrict__ src, const int* __restrict__ dst,
    const float* __restrict__ dinv, const float* __restrict__ t,
    float* __restrict__ agg) {
  const int lane = threadIdx.x & 31;
  const int warp = (blockIdx.x * blockDim.x + threadIdx.x) >> 5;
  const int nwarps = (gridDim.x * blockDim.x) >> 5;
  for (int e = warp; e < NEDGES; e += nwarps) {
    const int epf = e + 8 * nwarps;
    if (epf < NEDGES) {  // prefetch index stream ahead (global_prefetch_b8)
      __builtin_prefetch(&src[epf], 0, 0);
      __builtin_prefetch(&dst[epf], 0, 0);
    }
    const int s = src[e];
    const int d = dst[e];
    const float norm = dinv[s] * dinv[d];
    const float v = t[(size_t)s * 32 + lane] * norm;
    atomicAdd(&agg[(size_t)d * 32 + lane], v);
  }
}

// h = relu(agg + bias)
__global__ void __launch_bounds__(256) finalize_kernel(const float* __restrict__ agg,
                                                       const float* __restrict__ b,
                                                       float* __restrict__ h) {
  int i = blockIdx.x * blockDim.x + threadIdx.x;
  if (i < NNODES * 32) {
    float v = agg[i] + b[i & 31];
    h[i] = v > 0.f ? v : 0.f;
  }
}

// ---------------- predictor: out = tanh(tanh(h@Wp1+bp1)@wp2+bp2) + priors ----

__global__ void __launch_bounds__(256) predictor_kernel(
    const float* __restrict__ h,
    const __bf16* __restrict__ pp1, const float* __restrict__ bp1,
    const float* __restrict__ wp2, const float* __restrict__ bp2,
    const float* __restrict__ priors, float* __restrict__ out) {
  __shared__ float lds[8][16 * 32];
  const int wave = threadIdx.x >> 5;
  const int lane = threadIdx.x & 31;
  const int tile = blockIdx.x * 8 + wave;
  if (tile >= NTILES) return;
  const int row0 = tile * 16;
  const int n = lane & 15;
  const int mbase = (lane & 16) ? 8 : 0;

  v16bf a = load_a_f32(h, 32, row0, 0);
#pragma unroll
  for (int ct = 0; ct < 2; ++ct) {
    v16bf b = load_b_packed(pp1, ct);
    v8f acc = zero_v8f();
    acc = wmma_bf16(a, b, acc);
    const float bias = bp1[ct * 16 + n];
#pragma unroll
    for (int g = 0; g < 8; ++g)
      lds[wave][(mbase + g) * 32 + ct * 16 + n] = tanhf(acc[g] + bias);
  }

  // second layer: 32 -> 1, lanes 0..15 each handle one row
  if (lane < 16) {
    float s = 0.f;
#pragma unroll
    for (int k = 0; k < 32; ++k) s += lds[wave][lane * 32 + k] * wp2[k];
    out[row0 + lane] = tanhf(s + bp2[0]) + priors[row0 + lane];
  }
}

// ---------------- launch ----------------

extern "C" void kernel_launch(void* const* d_in, const int* in_sizes, int n_in,
                              void* d_out, int out_size, void* d_ws, size_t ws_size,
                              hipStream_t stream) {
  (void)in_sizes; (void)n_in; (void)out_size; (void)ws_size;
  const float* x      = (const float*)d_in[0];
  const int*   ei     = (const int*)d_in[1];
  const float* priors = (const float*)d_in[2];
  const float* we1 = (const float*)d_in[3];
  const float* be1 = (const float*)d_in[4];
  const float* we2 = (const float*)d_in[5];
  const float* be2 = (const float*)d_in[6];
  const float* wg1 = (const float*)d_in[7];
  const float* bg1 = (const float*)d_in[8];
  const float* wg2 = (const float*)d_in[9];
  const float* bg2 = (const float*)d_in[10];
  const float* wg3 = (const float*)d_in[11];
  const float* bg3 = (const float*)d_in[12];
  const float* wp1 = (const float*)d_in[13];
  const float* bp1 = (const float*)d_in[14];
  const float* wp2 = (const float*)d_in[15];
  const float* bp2 = (const float*)d_in[16];
  float* out = (float*)d_out;

  float* ws   = (float*)d_ws;
  float* h    = ws;                           // N*32
  float* t    = h   + (size_t)NNODES * 32;    // N*32
  float* agg  = t   + (size_t)NNODES * 32;    // N*32
  float* deg  = agg + (size_t)NNODES * 32;    // N
  float* dinv = deg + NNODES;                 // N
  // bf16 packed-weight pool (starts at a 32B-aligned offset)
  __bf16* pw1 = (__bf16*)(dinv + NNODES);     // 32 frags * 512 = 16384
  __bf16* pw2 = pw1 + 32 * 512;               // 4 frags  = 2048
  __bf16* pg1 = pw2 + 4 * 512;                // 2 frags  = 1024
  __bf16* pg2 = pg1 + 2 * 512;
  __bf16* pg3 = pg2 + 2 * 512;
  __bf16* pp1 = pg3 + 2 * 512;

  const int* src = ei;
  const int* dst = ei + NEDGES;

  const int nthreads  = 256;
  const int tblocks   = (NTILES + 7) / 8;             // 8 waves/block
  const int nblocksN  = (NNODES + 255) / 256;
  const int nblocksNF = (NNODES * 32 + 255) / 256;

  // pack weights into WMMA B-fragment order (re-done every call: deterministic)
  pack_b_kernel<<<4, nthreads, 0, stream>>>(we1, 256, 64, pw1);
  pack_b_kernel<<<1, nthreads, 0, stream>>>(we2, 64, 32, pw2);
  pack_b_kernel<<<1, nthreads, 0, stream>>>(wg1, 32, 32, pg1);
  pack_b_kernel<<<1, nthreads, 0, stream>>>(wg2, 32, 32, pg2);
  pack_b_kernel<<<1, nthreads, 0, stream>>>(wg3, 32, 32, pg3);
  pack_b_kernel<<<1, nthreads, 0, stream>>>(wp1, 32, 32, pp1);

  deg_init_kernel<<<nblocksN, nthreads, 0, stream>>>(deg);
  deg_accum_kernel<<<2048, nthreads, 0, stream>>>(dst, deg);
  dinv_kernel<<<nblocksN, nthreads, 0, stream>>>(deg, dinv);

  encoder_kernel<<<tblocks, nthreads, 0, stream>>>(x, pw1, be1, pw2, be2, h);

  const __bf16* PG[3] = {pg1, pg2, pg3};
  const float*  BG[3] = {bg1, bg2, bg3};
  for (int l = 0; l < 3; ++l) {
    gcn_mm_kernel<<<tblocks, nthreads, 0, stream>>>(h, PG[l], dinv, t, agg);
    edge_scatter_kernel<<<8192, nthreads, 0, stream>>>(src, dst, dinv, t, agg);
    finalize_kernel<<<nblocksNF, nthreads, 0, stream>>>(agg, BG[l], h);
  }

  predictor_kernel<<<tblocks, nthreads, 0, stream>>>(h, bp1 ? pp1 : pp1, bp1, wp2, bp2, priors, out);
}
